// AttentionMechanism_3530463117322
// MI455X (gfx1250) — compile-verified, matrix operand order checked by reading, not run on hardware
//
#include <hip/hip_runtime.h>

typedef __attribute__((ext_vector_type(16))) _Float16 v16h;
typedef __attribute__((ext_vector_type(8)))  _Float16 v8h;
typedef __attribute__((ext_vector_type(8)))  float    v8f;
typedef __attribute__((ext_vector_type(4)))  int      v4i;

#if defined(__has_builtin)
# if __has_builtin(__builtin_amdgcn_global_load_async_to_lds_b128) && \
     __has_builtin(__builtin_amdgcn_s_wait_asynccnt)
#  define ATHENA_ASYNC_LDS 1
# endif
#endif
#ifndef ATHENA_ASYNC_LDS
# define ATHENA_ASYNC_LDS 0
#endif

#if ATHENA_ASYNC_LDS
typedef __attribute__((address_space(1))) v4i* as1_v4i;
typedef __attribute__((address_space(3))) v4i* as3_v4i;
#endif

__device__ __forceinline__ v16h load_frag(const _Float16* p0, const _Float16* p1) {
    v8h lo = *(const v8h*)p0;
    v8h hi = *(const v8h*)p1;
    return __builtin_shufflevector(lo, hi, 0,1,2,3,4,5,6,7,8,9,10,11,12,13,14,15);
}

#define WMMA_F32_F16(a, b, c) \
    __builtin_amdgcn_wmma_f32_16x16x32_f16(false, (a), false, (b), (short)0, (c), false, false)

// ---------------------------------------------------------------------------
// Projection GEMM: Y[m][n] = (f16)( X[m][:] . W[n][:] + bias[n] )
// X: [M,K] f32 row-major, W: [N,K] f32 row-major (torch Linear weight layout).
// Block: 256 thr (8 waves), tile 128(M) x 128(N), K-step 32.
// Wave grid: 2 (M, 64 rows each) x 4 (N, 32 cols each); 8 WMMAs/wave/K-step.
// ---------------------------------------------------------------------------
#define PT_K 32
#define PT_PAD 8

__global__ __launch_bounds__(256) void proj_kernel(
    const float* __restrict__ X, const float* __restrict__ W,
    const float* __restrict__ bias, _Float16* __restrict__ Y,
    int M, int K, int N)
{
    __shared__ _Float16 lA[128][PT_K + PT_PAD];
    __shared__ _Float16 lB[128][PT_K + PT_PAD];

    const int tid    = threadIdx.x;
    const int wid    = tid >> 5;
    const int lane   = tid & 31;
    const int laneHi = lane >> 4;
    const int l15    = lane & 15;

    const int m0 = blockIdx.x * 128;
    const int n0 = blockIdx.y * 128;
    const int wm = (wid & 1) * 64;   // wave M offset within tile
    const int wn = (wid >> 1) * 32;  // wave N offset within tile

    v8f acc[4][2];
#pragma unroll
    for (int i = 0; i < 4; ++i)
#pragma unroll
        for (int j = 0; j < 2; ++j)
#pragma unroll
            for (int e = 0; e < 8; ++e) acc[i][j][e] = 0.0f;

    // LDS-fill coordinates: each thread converts 16 f32 -> 16 f16 of one row
    const int fr = tid >> 1;
    const int fc = (tid & 1) * 16;

    for (int k0 = 0; k0 < K; k0 += PT_K) {
        float a4[16], b4[16];
        {
            const float4* ga = (const float4*)(X + (size_t)(m0 + fr) * K + k0 + fc);
            const float4* gb = (const float4*)(W + (size_t)(n0 + fr) * K + k0 + fc);
#pragma unroll
            for (int i = 0; i < 4; ++i) {
                float4 a = ga[i]; float4 b = gb[i];
                a4[4*i+0]=a.x; a4[4*i+1]=a.y; a4[4*i+2]=a.z; a4[4*i+3]=a.w;
                b4[4*i+0]=b.x; b4[4*i+1]=b.y; b4[4*i+2]=b.z; b4[4*i+3]=b.w;
            }
        }
        __syncthreads();   // previous iteration's readers done
        {
            v8h h0, h1, g0, g1;
#pragma unroll
            for (int i = 0; i < 8; ++i) {
                h0[i] = (_Float16)a4[i]; h1[i] = (_Float16)a4[i+8];
                g0[i] = (_Float16)b4[i]; g1[i] = (_Float16)b4[i+8];
            }
            *(v8h*)&lA[fr][fc]     = h0;  *(v8h*)&lA[fr][fc + 8] = h1;
            *(v8h*)&lB[fr][fc]     = g0;  *(v8h*)&lB[fr][fc + 8] = g1;
        }
        __syncthreads();

        v16h af[4];
#pragma unroll
        for (int ms = 0; ms < 4; ++ms) {
            const int row = wm + ms * 16 + l15;
            af[ms] = load_frag(&lA[row][laneHi * 8], &lA[row][16 + laneHi * 8]);
        }
#pragma unroll
        for (int ns = 0; ns < 2; ++ns) {
            const int col = wn + ns * 16 + l15;
            v16h bf = load_frag(&lB[col][laneHi * 8], &lB[col][16 + laneHi * 8]);
#pragma unroll
            for (int ms = 0; ms < 4; ++ms)
                acc[ms][ns] = WMMA_F32_F16(af[ms], bf, acc[ms][ns]);
        }
    }

    // Epilogue: C layout: vgpr j, lanes0-15 -> M = j, lanes16-31 -> M = j+8
#pragma unroll
    for (int ns = 0; ns < 2; ++ns) {
        const int n = n0 + wn + ns * 16 + l15;
        const float bv = bias[n];
#pragma unroll
        for (int ms = 0; ms < 4; ++ms) {
            const int mb = m0 + wm + ms * 16 + laneHi * 8;
#pragma unroll
            for (int j = 0; j < 8; ++j)
                Y[(size_t)(mb + j) * N + n] = (_Float16)(acc[ms][ns][j] + bv);
        }
    }
}

// ---------------------------------------------------------------------------
// docs [B, ND, D] f32  ->  docsT [B, D, ND] f16   (32x32 LDS tile transpose)
// ---------------------------------------------------------------------------
__global__ __launch_bounds__(256) void transpose_docs_kernel(
    const float* __restrict__ docs, _Float16* __restrict__ docsT, int ND, int D)
{
    __shared__ _Float16 t[32][33];
    const int b  = blockIdx.z;
    const int n0 = blockIdx.x * 32;   // doc tile
    const int d0 = blockIdx.y * 32;   // feature tile
    const float*    src = docs  + (size_t)b * ND * D;
    _Float16*       dst = docsT + (size_t)b * ND * D;
    const int tx = threadIdx.x & 31;
    const int ty = threadIdx.x >> 5;  // 0..7
#pragma unroll
    for (int i = 0; i < 32; i += 8)
        t[ty + i][tx] = (_Float16)src[(size_t)(n0 + ty + i) * D + d0 + tx];
    __syncthreads();
#pragma unroll
    for (int i = 0; i < 32; i += 8)
        dst[(size_t)(d0 + ty + i) * ND + n0 + tx] = t[tx][ty + i];
}

// ---------------------------------------------------------------------------
// Fused attention: one 16-query tile per workgroup (8 waves).
//   scores  : wave w computes S(16q x 16docs) for docs [nb*128+16w, +16)
//   softmax : online (running max m, running sum l) across 32 blocks of 128
//   output  : wave w accumulates 16 x 128 slice (cols [128w,128w+128)) of out
// ---------------------------------------------------------------------------
__global__ __launch_bounds__(256) void attn_kernel(
    const _Float16* __restrict__ q16,    // [B, LQ, D]
    const _Float16* __restrict__ k16,    // [B, ND, D]
    const _Float16* __restrict__ dTg,    // [B, D, ND]
    float* __restrict__ out,             // [B, LQ, D]
    int LQ, int ND, int D)
{
    __shared__ _Float16 qs[16][1024 + 8];
    __shared__ _Float16 ps[16][128 + 8];
    __shared__ float wmax[8][16];
    __shared__ float wsum[8][16];
    __shared__ float row_m[16], row_scale[16], row_l[16];

    const int tid    = threadIdx.x;
    const int wid    = tid >> 5;
    const int lane   = tid & 31;
    const int laneHi = lane >> 4;
    const int l15    = lane & 15;

    const int b  = blockIdx.y;
    const int m0 = blockIdx.x * 16;

    const _Float16* qg = q16 + ((size_t)b * LQ + m0) * D;
    const _Float16* kg = k16 + (size_t)b * ND * D;
    const _Float16* dT = dTg + (size_t)b * (size_t)ND * D;

    // Stage 16 x 1024 query tile into LDS (each thread moves 64 halves).
    // Preferred path: gfx1250 async global->LDS DMA, tracked by ASYNCcnt.
    {
        const int row = tid >> 4;
        const int cb  = (tid & 15) * 64;
        const _Float16* src = qg + (size_t)row * D + cb;
        _Float16*       dst = &qs[row][cb];
#if ATHENA_ASYNC_LDS
#pragma unroll
        for (int i = 0; i < 8; ++i)
            __builtin_amdgcn_global_load_async_to_lds_b128(
                (as1_v4i)(v4i*)(src + i * 8), (as3_v4i)(v4i*)(dst + i * 8), 0, 0);
        __builtin_amdgcn_s_wait_asynccnt(0);
#else
        const v8h* s = (const v8h*)src;
        v8h* d = (v8h*)dst;
#pragma unroll
        for (int i = 0; i < 8; ++i) d[i] = s[i];
#endif
    }
    if (tid < 16) { row_m[tid] = -3.0e38f; row_l[tid] = 0.0f; }
    __syncthreads();

    v8f o[8];
#pragma unroll
    for (int t = 0; t < 8; ++t)
#pragma unroll
        for (int e = 0; e < 8; ++e) o[t][e] = 0.0f;

    const int nblocks = ND / 128;
    for (int nb = 0; nb < nblocks; ++nb) {
        const int dbase = nb * 128;

        // ---- scores: S = q(16 x D) . k_docs(16 x D)^T, this wave's docs
        v8f s;
#pragma unroll
        for (int e = 0; e < 8; ++e) s[e] = 0.0f;
        const _Float16* kw = kg + (size_t)(dbase + wid * 16 + l15) * D;
        // prefetch this lane's doc row for the NEXT block (k16 is L2-resident)
        if (dbase + 128 < ND)
            __builtin_prefetch(kw + (size_t)128 * D, 0, 1);
#pragma unroll 4
        for (int kk = 0; kk < 1024; kk += 32) {
            v16h af = load_frag(&qs[l15][kk + laneHi * 8], &qs[l15][kk + 16 + laneHi * 8]);
            v16h bf = load_frag(kw + kk + laneHi * 8,       kw + kk + 16 + laneHi * 8);
            s = WMMA_F32_F16(af, bf, s);
        }

        // ---- per-row max over this wave's 16 docs (reduce across half-wave)
#pragma unroll
        for (int j = 0; j < 8; ++j) {
            float v = s[j];
#pragma unroll
            for (int off = 8; off >= 1; off >>= 1)
                v = fmaxf(v, __shfl_xor(v, off, 16));
            if (l15 == 0) wmax[wid][j + laneHi * 8] = v;
        }
        __syncthreads();
        if (tid < 16) {
            float bm = wmax[0][tid];
#pragma unroll
            for (int w = 1; w < 8; ++w) bm = fmaxf(bm, wmax[w][tid]);
            const float om = row_m[tid];
            const float nm = fmaxf(om, bm);
            row_m[tid]     = nm;
            row_scale[tid] = __expf(om - nm);
        }
        __syncthreads();

        // ---- P = exp(S - m): stage to LDS, per-wave row sums
#pragma unroll
        for (int j = 0; j < 8; ++j) {
            const int row = j + laneHi * 8;
            const float p = __expf(s[j] - row_m[row]);
            float v = p;
#pragma unroll
            for (int off = 8; off >= 1; off >>= 1)
                v += __shfl_xor(v, off, 16);
            if (l15 == 0) wsum[wid][row] = v;
            ps[row][wid * 16 + l15] = (_Float16)p;
        }
        // rescale output accumulators by exp(m_old - m_new)
        {
            float sc[8];
#pragma unroll
            for (int j = 0; j < 8; ++j) sc[j] = row_scale[j + laneHi * 8];
#pragma unroll
            for (int t = 0; t < 8; ++t)
#pragma unroll
                for (int j = 0; j < 8; ++j) o[t][j] *= sc[j];
        }
        __syncthreads();
        if (tid < 16) {
            float l = row_l[tid] * row_scale[tid];
#pragma unroll
            for (int w = 0; w < 8; ++w) l += wsum[w][tid];
            row_l[tid] = l;
        }

        // ---- out += P(16 x 128) . docs_block(128 x 128-col slice)
        const int cchunk = wid * 128;
#pragma unroll
        for (int kk = 0; kk < 128; kk += 32) {
            v16h af = load_frag(&ps[l15][kk + laneHi * 8], &ps[l15][kk + 16 + laneHi * 8]);
#pragma unroll
            for (int t = 0; t < 8; ++t) {
                const int col = cchunk + t * 16 + l15;
                const _Float16* dp = dT + (size_t)col * ND + dbase + kk;
                v16h bf = load_frag(dp + laneHi * 8, dp + 16 + laneHi * 8);
                o[t] = WMMA_F32_F16(af, bf, o[t]);
            }
        }
    }

    __syncthreads();
    float linv[8];
#pragma unroll
    for (int j = 0; j < 8; ++j) linv[j] = 1.0f / row_l[j + laneHi * 8];
    float* og = out + ((size_t)b * LQ + m0) * D;
#pragma unroll
    for (int t = 0; t < 8; ++t) {
        const int col = wid * 128 + t * 16 + l15;
#pragma unroll
        for (int j = 0; j < 8; ++j) {
            const int row = j + laneHi * 8;
            og[(size_t)row * D + col] = o[t][j] * linv[j];
        }
    }
}

// ---------------------------------------------------------------------------
extern "C" void kernel_launch(void* const* d_in, const int* in_sizes, int n_in,
                              void* d_out, int out_size, void* d_ws, size_t ws_size,
                              hipStream_t stream) {
    const float* query = (const float*)d_in[0];  // [B, LQ, D]
    const float* docs  = (const float*)d_in[1];  // [B, ND, D]
    const float* Wq    = (const float*)d_in[2];  // [D, D]
    const float* bq    = (const float*)d_in[3];  // [D]
    const float* Wk    = (const float*)d_in[4];  // [D, D]
    const float* bk    = (const float*)d_in[5];  // [D]
    float* outp = (float*)d_out;

    const int B = 4, LQ = 1024, ND = 4096, D = 1024;

    _Float16* q16 = (_Float16*)d_ws;                 //  8 MB
    _Float16* k16 = q16 + (size_t)B * LQ * D;        // 32 MB
    _Float16* dT  = k16 + (size_t)B * ND * D;        // 32 MB

    dim3 blk(256);
    // q = query @ Wq^T + bq  (batch folded into M)
    proj_kernel<<<dim3(B * LQ / 128, D / 128), blk, 0, stream>>>(query, Wq, bq, q16, B * LQ, D, D);
    // k = documents @ Wk^T + bk
    proj_kernel<<<dim3(B * ND / 128, D / 128), blk, 0, stream>>>(docs, Wk, bk, k16, B * ND, D, D);
    // docsT[b][d][n] = (f16) docs[b][n][d]
    transpose_docs_kernel<<<dim3(ND / 32, D / 32, B), blk, 0, stream>>>(docs, dT, ND, D);
    // fused scores + softmax + weighted sum
    attn_kernel<<<dim3(LQ / 16, B), blk, 0, stream>>>(q16, k16, dT, outp, LQ, ND, D);
}
